// GNNRewriteDiscriminator_42133629173808
// MI455X (gfx1250) — matrix operand order repaired and checked by reading, not compiled, not run on hardware
//
#include <hip/hip_runtime.h>
#include <hip/hip_bf16.h>
#include <hip/hip_fp16.h>

typedef __attribute__((ext_vector_type(16))) _Float16 v16h;
typedef __attribute__((ext_vector_type(8)))  float    v8f;

#define TPB 256

// ---------------- node/degree helpers ----------------

__global__ void k_set1(float* __restrict__ p, int n) {
  int i = blockIdx.x * blockDim.x + threadIdx.x;
  if (i < n) p[i] = 1.0f;
}

__global__ void k_degree(const int* __restrict__ dst, float* __restrict__ deg, int E_) {
  int e = blockIdx.x * blockDim.x + threadIdx.x;
  if (e < E_) unsafeAtomicAdd(&deg[dst[e]], 1.0f);
}

__global__ void k_dinv(float* __restrict__ d, int n) {
  int i = blockIdx.x * blockDim.x + threadIdx.x;
  if (i < n) d[i] = rsqrtf(fmaxf(d[i], 1.0f));
}

// x = emb[gate_ids] + params*pW + pb  -> f16 [nPad, 32] (cols 16..31 zero pad for K=32 WMMA)
__global__ void k_nodefeat(const int* __restrict__ gids, const float* __restrict__ params,
                           const float* __restrict__ emb, const float* __restrict__ pW,
                           const float* __restrict__ pb, _Float16* __restrict__ xA,
                           int N_, int nPad) {
  int i = blockIdx.x * blockDim.x + threadIdx.x;
  if (i >= nPad) return;
  _Float16* row = xA + (size_t)i * 32;
  if (i < N_) {
    int gid = gids[i];
    float p = params[i];
#pragma unroll
    for (int c = 0; c < 16; ++c)
      row[c] = (_Float16)(emb[gid * 16 + c] + p * pW[c] + pb[c]);
#pragma unroll
    for (int c = 16; c < 32; ++c) row[c] = (_Float16)0.0f;
  } else {
#pragma unroll
    for (int c = 0; c < 32; ++c) row[c] = (_Float16)0.0f;
  }
}

// Pre-swizzle W [Kdim,32] (f32) into per-lane WMMA B fragments, f16, zero-padded K.
// Layout: Bfrag[half][lane][e], half = output-column half (N 0-15 / 16-31),
// lane = g*16+r (g: lane half, r: column within half), e: K = g*8+e (e<8), 16+g*8+(e-8) (e>=8).
__global__ void k_prep_B(const float* __restrict__ W, _Float16* __restrict__ Bfrag, int Kdim) {
  int idx = threadIdx.x;          // 0..63 : {half, lane}
  if (idx >= 64) return;
  int half = idx >> 5;
  int lane = idx & 31;
  int g = lane >> 4, r = lane & 15;
  _Float16* out = Bfrag + (size_t)(half * 32 + lane) * 16;
#pragma unroll
  for (int e = 0; e < 16; ++e) {
    int K = (e < 8) ? (g * 8 + e) : (16 + g * 8 + (e - 8));
    out[e] = (_Float16)((K < Kdim) ? W[K * 32 + half * 16 + r] : 0.0f);
  }
}

// ---------------- WMMA dense matmul: H[N,32] = A[N,32(f16)] @ W[Kdim,32] ----------------
// One wave per 16-row tile; two v_wmma_f32_16x16x32_f16 per tile (output col halves).
// A fragment per lane = two contiguous 16B chunks of row r: K = g*8+[0..7] and g*8+[16..23].
__global__ void k_wmma_mm(const _Float16* __restrict__ A, const _Float16* __restrict__ Bfrag,
                          float* __restrict__ H, int nTiles) {
  int wave = (blockIdx.x * blockDim.x + threadIdx.x) >> 5;
  int lane = threadIdx.x & 31;
  if (wave >= nTiles) return;
  int g = lane >> 4;
  int r = lane & 15;

  const _Float16* arow = A + (size_t)(wave * 16 + r) * 32 + g * 8;
  union { v16h v; float4 f4[2]; } au;
  au.f4[0] = *(const float4*)(arow);        // elements 0..7  -> K = g*8 + 0..7
  au.f4[1] = *(const float4*)(arow + 16);   // elements 8..15 -> K = g*8 + 16..23

  v16h b0 = *(const v16h*)(Bfrag + (size_t)lane * 16);
  v16h b1 = *(const v16h*)(Bfrag + (size_t)(32 + lane) * 16);

  v8f c0 = {}, c1 = {};
  c0 = __builtin_amdgcn_wmma_f32_16x16x32_f16(false, au.v, false, b0, (short)0, c0, false, false);
  c1 = __builtin_amdgcn_wmma_f32_16x16x32_f16(false, au.v, false, b1, (short)0, c1, false, false);

  // D layout: VGPR rr -> M=rr (lanes 0-15) / M=rr+8 (lanes 16-31), N=r
  float* hrow = H + (size_t)wave * 16 * 32;
#pragma unroll
  for (int rr = 0; rr < 8; ++rr) {
    int M = rr + g * 8;
    hrow[M * 32 + r]      = c0[rr];
    hrow[M * 32 + 16 + r] = c1[rr];
  }
}

// ---------------- edge scatter: acc[dst] += h[src] * dinv[src]*dinv[dst] ----------------
// 8 threads per edge, float4 chunks: consecutive threads touch contiguous bytes of one row.
__global__ void k_edge_scatter(const int* __restrict__ src, const int* __restrict__ dst,
                               const float* __restrict__ dinv, const float* __restrict__ H,
                               float* __restrict__ acc, int E_) {
  int idx = blockIdx.x * blockDim.x + threadIdx.x;
  if (idx >= E_ * 8) return;
  int e = idx >> 3;
  int c = idx & 7;
  int s = src[e], d = dst[e];
  float norm = dinv[s] * dinv[d];
  float4 hv = *(const float4*)(H + (size_t)s * 32 + (c << 2));
  float* a = acc + (size_t)d * 32 + (c << 2);
  unsafeAtomicAdd(a + 0, hv.x * norm);
  unsafeAtomicAdd(a + 1, hv.y * norm);
  unsafeAtomicAdd(a + 2, hv.z * norm);
  unsafeAtomicAdd(a + 3, hv.w * norm);
}

// out = relu(acc + h*dinv^2 (self loop) + bias) -> f16 A buffer for next conv
__global__ void k_finalize(const float* __restrict__ acc, const float* __restrict__ H,
                           const float* __restrict__ dinv, const float* __restrict__ bias,
                           _Float16* __restrict__ xA, int nPad) {
  int i = blockIdx.x * blockDim.x + threadIdx.x;
  if (i >= nPad) return;
  float sn = dinv[i] * dinv[i];
#pragma unroll
  for (int c = 0; c < 32; ++c) {
    float v = acc[(size_t)i * 32 + c] + H[(size_t)i * 32 + c] * sn + bias[c];
    xA[(size_t)i * 32 + c] = (_Float16)fmaxf(v, 0.0f);
  }
}

// conv2 finalize + segment-sum pooling via atomics; cnt once per node
__global__ void k_finalize_pool(const float* __restrict__ acc, const float* __restrict__ H,
                                const float* __restrict__ dinv, const float* __restrict__ bias,
                                const int* __restrict__ batch, float* __restrict__ pool,
                                float* __restrict__ cnt, int side, int N_) {
  int i = blockIdx.x * blockDim.x + threadIdx.x;
  if (i >= N_) return;
  float sn = dinv[i] * dinv[i];
  int b = batch[i];
  float* prow = pool + (size_t)b * 64 + side * 32;
#pragma unroll
  for (int c = 0; c < 32; ++c) {
    float v = acc[(size_t)i * 32 + c] + H[(size_t)i * 32 + c] * sn + bias[c];
    unsafeAtomicAdd(prow + c, fmaxf(v, 0.0f));
  }
  unsafeAtomicAdd(cnt + b, 1.0f);
}

// final head: mean-pool division + [G,64]@[64,32] relu @[32,1]
__global__ void k_final_mlp(const float* __restrict__ pool, const float* __restrict__ cnt,
                            const float* __restrict__ fW1, const float* __restrict__ fb1,
                            const float* __restrict__ fW2, const float* __restrict__ fb2,
                            float* __restrict__ out, int G_) {
  int gi = blockIdx.x * blockDim.x + threadIdx.x;
  if (gi >= G_) return;
  float hcat[64];
  float c0 = fmaxf(cnt[gi], 1.0f);
  float c1 = fmaxf(cnt[G_ + gi], 1.0f);
#pragma unroll
  for (int c = 0; c < 32; ++c) hcat[c] = pool[(size_t)gi * 64 + c] / c0;
#pragma unroll
  for (int c = 0; c < 32; ++c) hcat[32 + c] = pool[(size_t)gi * 64 + 32 + c] / c1;
  float o = fb2[0];
  for (int j = 0; j < 32; ++j) {
    float s = fb1[j];
#pragma unroll
    for (int c = 0; c < 64; ++c) s += hcat[c] * fW1[c * 32 + j];
    o += fmaxf(s, 0.0f) * fW2[j];
  }
  out[gi] = o;
}

extern "C" void kernel_launch(void* const* d_in, const int* in_sizes, int n_in,
                              void* d_out, int out_size, void* d_ws, size_t ws_size,
                              hipStream_t stream) {
  const int*   lg  = (const int*)d_in[0];
  const float* lp  = (const float*)d_in[1];
  const int*   le  = (const int*)d_in[2];
  const int*   lb  = (const int*)d_in[3];
  const int*   rg  = (const int*)d_in[4];
  const float* rp  = (const float*)d_in[5];
  const int*   re  = (const int*)d_in[6];
  const int*   rb  = (const int*)d_in[7];
  const float* emb = (const float*)d_in[8];
  const float* pW  = (const float*)d_in[9];
  const float* pb  = (const float*)d_in[10];
  const float* W1  = (const float*)d_in[11];
  const float* b1  = (const float*)d_in[12];
  const float* W2  = (const float*)d_in[13];
  const float* b2  = (const float*)d_in[14];
  const float* fW1 = (const float*)d_in[15];
  const float* fb1 = (const float*)d_in[16];
  const float* fW2 = (const float*)d_in[17];
  const float* fb2 = (const float*)d_in[18];

  const int N_ = in_sizes[0];
  const int E_ = in_sizes[2] / 2;
  const int G_ = out_size;
  const int nPad   = ((N_ + 15) / 16) * 16;
  const int nTiles = nPad / 16;

  // carve workspace (shared across the two sequentially-processed graphs)
  char* ws = (char*)d_ws;
  size_t off = 0;
  auto carve = [&](size_t bytes) -> char* {
    char* p = ws + off;
    off = (off + bytes + 255) & ~(size_t)255;
    return p;
  };
  float*    dinv = (float*)   carve((size_t)nPad * sizeof(float));
  _Float16* xA   = (_Float16*)carve((size_t)nPad * 32 * sizeof(_Float16));
  float*    h    = (float*)   carve((size_t)nPad * 32 * sizeof(float));
  float*    acc  = (float*)   carve((size_t)nPad * 32 * sizeof(float));
  float*    pool = (float*)   carve((size_t)G_ * 64 * sizeof(float));
  float*    cnt  = (float*)   carve((size_t)2 * G_ * sizeof(float));
  _Float16* Bf1  = (_Float16*)carve((size_t)2 * 32 * 16 * sizeof(_Float16));
  _Float16* Bf2  = (_Float16*)carve((size_t)2 * 32 * 16 * sizeof(_Float16));
  (void)n_in; (void)ws_size;

  const int nb_n  = (nPad + TPB - 1) / TPB;
  const int nb_N  = (N_ + TPB - 1) / TPB;
  const int nb_e  = (E_ + TPB - 1) / TPB;
  const int nb_e8 = (E_ * 8 + TPB - 1) / TPB;
  const int nb_w  = (nTiles * 32 + TPB - 1) / TPB;

  hipMemsetAsync(pool, 0, (size_t)G_ * 64 * sizeof(float), stream);
  hipMemsetAsync(cnt, 0, (size_t)2 * G_ * sizeof(float), stream);
  k_prep_B<<<1, 64, 0, stream>>>(W1, Bf1, 16);
  k_prep_B<<<1, 64, 0, stream>>>(W2, Bf2, 32);

  for (int side = 0; side < 2; ++side) {
    const int*   gids   = side ? rg : lg;
    const float* params = side ? rp : lp;
    const int*   ei     = side ? re : le;
    const int*   batch  = side ? rb : lb;
    const int* srcp = ei;        // edge_index[0]
    const int* dstp = ei + E_;   // edge_index[1]

    k_set1<<<nb_n, TPB, 0, stream>>>(dinv, nPad);                 // self-loop degree
    k_degree<<<nb_e, TPB, 0, stream>>>(dstp, dinv, E_);
    k_dinv<<<nb_n, TPB, 0, stream>>>(dinv, nPad);
    k_nodefeat<<<nb_n, TPB, 0, stream>>>(gids, params, emb, pW, pb, xA, N_, nPad);

    hipMemsetAsync(acc, 0, (size_t)nPad * 32 * sizeof(float), stream);
    k_wmma_mm<<<nb_w, TPB, 0, stream>>>(xA, Bf1, h, nTiles);
    k_edge_scatter<<<nb_e8, TPB, 0, stream>>>(srcp, dstp, dinv, h, acc, E_);
    k_finalize<<<nb_n, TPB, 0, stream>>>(acc, h, dinv, b1, xA, nPad);

    hipMemsetAsync(acc, 0, (size_t)nPad * 32 * sizeof(float), stream);
    k_wmma_mm<<<nb_w, TPB, 0, stream>>>(xA, Bf2, h, nTiles);
    k_edge_scatter<<<nb_e8, TPB, 0, stream>>>(srcp, dstp, dinv, h, acc, E_);
    k_finalize_pool<<<nb_N, TPB, 0, stream>>>(acc, h, dinv, b2, batch, pool,
                                              cnt + side * G_, side, N_);
  }

  k_final_mlp<<<(G_ + TPB - 1) / TPB, TPB, 0, stream>>>(pool, cnt, fW1, fb1, fW2, fb2,
                                                        (float*)d_out, G_);
}
